// DRN_78520592105913
// MI455X (gfx1250) — matrix operand order repaired and checked by reading, not compiled, not run on hardware
//
#include <hip/hip_runtime.h>

// ---------------------------------------------------------------------------
// Fused DRN forward for gfx1250 (MI455X, wave32, WMMA), round 4 (final tune).
//
//   out = softmax_l( sum_k log( exp(-W[j,k]*d2) @ x[i,k,:] ) + bias(j,l) )
//
//  * T_k[l,m] = f(|m-l|): all 64 per-k diagonals precomputed once in LDS,
//    stored SIGNED and 4-way shift-replicated => each lane's A-fragment is
//    two runs of 16 consecutive 8B-aligned halfs -> 8x ds_load_b64 per k.
//  * sum_k log Pw == log prod_k Pw (Pw in [0.886,1.129], product f32-safe):
//    loop is med3-clip + v_pk_mul only; ONE raw v_log_f32/elem afterwards.
//  * log2 domain end-to-end (raw v_exp/v_log are base-2; softmax base-inv).
//  * x pre-converted to f16 in d_ws (B-fragment = 2x global_load_b128);
//    f32 fallback if ws too small.
//  * WMMAs ordered so consecutive ops share the A operand and alternate
//    accumulators (no back-to-back D->C chains).
//  * softmax LDS stage aliases the (dead) table: 68 KB total LDS.
// ---------------------------------------------------------------------------

typedef __attribute__((ext_vector_type(16))) _Float16 v16h;
typedef __attribute__((ext_vector_type(8)))  _Float16 v8h;
typedef __attribute__((ext_vector_type(4)))  _Float16 v4h;
typedef __attribute__((ext_vector_type(8)))  float    v8f;
typedef __attribute__((ext_vector_type(4)))  float    f32x4;

#define B_TOT 256
#define F     64            // F_in == F_out
#define Q     64            // ql == qu
#define L2E   1.44269504088896340736f
#define TSTR  136           // halfs per (copy,k) row; 136 % 4 == 0
#define NCPY  4             // shift-replicated copies (8B alignment classes)

// ---------------- pre-pass: x (f32) -> xh (f16), 8 elems/thread -------------
extern "C" __global__ __launch_bounds__(256)
void cvt_x_f16(const float* __restrict__ x, _Float16* __restrict__ xh)
{
    int g = blockIdx.x * 256 + threadIdx.x;          // 131072 groups of 8
    const f32x4* p = (const f32x4*)x + (size_t)g * 2;
    f32x4 a = p[0], b = p[1];
    v8h h;
    h[0] = (_Float16)a[0]; h[1] = (_Float16)a[1];
    h[2] = (_Float16)a[2]; h[3] = (_Float16)a[3];
    h[4] = (_Float16)b[0]; h[5] = (_Float16)b[1];
    h[6] = (_Float16)b[2]; h[7] = (_Float16)b[3];
    *(v8h*)(xh + (size_t)g * 8) = h;
}

// f32 fallback B-fragment load (16 contiguous floats -> v16h)
__device__ __forceinline__ v16h load_b_frag_f32(const float* __restrict__ p) {
    const f32x4* p4 = (const f32x4*)p;
    f32x4 q0 = p4[0], q1 = p4[1], q2 = p4[2], q3 = p4[3];
    v16h r;
#pragma unroll
    for (int e = 0; e < 4; ++e) {
        r[e]      = (_Float16)q0[e];
        r[e + 4]  = (_Float16)q1[e];
        r[e + 8]  = (_Float16)q2[e];
        r[e + 12] = (_Float16)q3[e];
    }
    return r;
}

template <bool USE_H>
__global__ __launch_bounds__(256)
void drn_fused(const float* __restrict__ x,      // (B, F, Q) f32
               const _Float16* __restrict__ xh,  // (B, F, Q) f16 (USE_H)
               const float* __restrict__ W,      // (F, F)
               const float* __restrict__ ba, const float* __restrict__ bq,
               const float* __restrict__ lama, const float* __restrict__ lamq,
               float* __restrict__ out)          // (B, F, Q)
{
    // 68 KB: signed diag tables; first 16 KB reused for softmax stage later.
    __shared__ __align__(16) unsigned char smem[NCPY * F * TSTR * 2];
    _Float16* tbl   = (_Float16*)smem;
    float*    lsmem = (float*)smem;

    const int j    = blockIdx.x;                // output feature
    const int i0   = blockIdx.y << 6;           // batch tile base
    const int tid  = threadIdx.x;
    const int wv   = tid >> 5;                  // wave 0..7
    const int ln   = tid & 31;
    const int lmod = ln & 15;
    const int hi   = ln >> 4;                   // lane half

    const int lblk = wv >> 1;                   // l rows [lblk*16,+16)
    const int ib0  = (wv & 1) << 1;             // i blocks ib0, ib0+1
    const int arow = (lblk << 4) + lmod;        // A-fragment row (= l)

    // ---- build signed, shift-replicated diag tables -----------------------
    // copy c, row k, displacement d stored at (c*F + k)*TSTR + 63 + c + d
    {
        int   k  = tid >> 2;                    // one k per thread, 16 t's
        int   t0 = (tid & 3) << 4;
        float c  = -W[j * F + k] * (L2E / 4096.0f);
#pragma unroll
        for (int u = 0; u < 16; ++u) {
            int      t = t0 + u;
            _Float16 v = (_Float16)__builtin_amdgcn_exp2f(c * (float)t * (float)t);
#pragma unroll
            for (int cc = 0; cc < NCPY; ++cc) {
                int base = (cc * F + k) * TSTR + 63 + cc;
                tbl[base + t] = v;
                tbl[base - t] = v;
            }
        }
    }

    // ---- per-lane table base: runs at +0,+32, each 16 halfs ---------------
    const int d0 = (hi << 4) - arow;            // first A element displacement
    const int c4 = (-(63 + d0)) & 3;            // copy making runs 8B-aligned
    const _Float16* tb = tbl + (c4 * F) * TSTR + 63 + c4 + d0;

    // ---- B-fragment base (col i = i0+ib*16+lmod, m = mb*32+hi*16) ---------
    const size_t cofs = ((size_t)(i0 + (ib0 << 4) + lmod) * F) * Q + (size_t)hi * 16;
    const float*    xc0f = x  + cofs;
    const _Float16* xc0h = xh + cofs;

    v8f prod0 = {1.f,1.f,1.f,1.f,1.f,1.f,1.f,1.f};  // prod_k Pw (i-block ib0)
    v8f prod1 = {1.f,1.f,1.f,1.f,1.f,1.f,1.f,1.f};  // prod_k Pw (i-block ib0+1)

    __syncthreads();                            // tables ready

#pragma unroll 2
    for (int k = 0; k < F; ++k) {
        // A fragments: 8 aligned ds_load_b64 (2 runs of 16 consecutive halfs)
        const _Float16* tr = tb + k * TSTR;
        v4h r0 = *(const v4h*)(tr);
        v4h r1 = *(const v4h*)(tr + 4);
        v4h r2 = *(const v4h*)(tr + 8);
        v4h r3 = *(const v4h*)(tr + 12);
        v4h r4 = *(const v4h*)(tr + 32);
        v4h r5 = *(const v4h*)(tr + 36);
        v4h r6 = *(const v4h*)(tr + 40);
        v4h r7 = *(const v4h*)(tr + 44);
        v16h a0, a1;
#pragma unroll
        for (int e = 0; e < 4; ++e) {
            a0[e] = r0[e]; a0[e + 4] = r1[e]; a0[e + 8] = r2[e]; a0[e + 12] = r3[e];
            a1[e] = r4[e]; a1[e + 4] = r5[e]; a1[e + 8] = r6[e]; a1[e + 12] = r7[e];
        }

        // B fragments
        v16h b00, b01, b10, b11;
        if (USE_H) {
            const _Float16* pk0 = xc0h + (size_t)k * Q;
            const _Float16* pk1 = pk0 + (size_t)16 * F * Q;   // +16 batches
            b00 = *(const v16h*)(pk0);
            b01 = *(const v16h*)(pk0 + 32);
            b10 = *(const v16h*)(pk1);
            b11 = *(const v16h*)(pk1 + 32);
            if (k + 1 < F) {
                __builtin_prefetch(pk0 + Q, 0, 3);
                __builtin_prefetch(pk1 + Q, 0, 3);
            }
        } else {
            const float* pk0 = xc0f + (size_t)k * Q;
            const float* pk1 = pk0 + (size_t)16 * F * Q;
            b00 = load_b_frag_f32(pk0);
            b01 = load_b_frag_f32(pk0 + 32);
            b10 = load_b_frag_f32(pk1);
            b11 = load_b_frag_f32(pk1 + 32);
        }

        // Pw tiles (16x16x64 = two 16x16x32 steps per tile). Order: share A
        // between consecutive WMMAs, alternate accumulators (no back-to-back
        // D->C dependency on the same accumulator chain).
        v8f p0 = {0.f,0.f,0.f,0.f,0.f,0.f,0.f,0.f};
        v8f p1 = {0.f,0.f,0.f,0.f,0.f,0.f,0.f,0.f};
        p0 = __builtin_amdgcn_wmma_f32_16x16x32_f16(false, a0, false, b00,
                                                    (short)0, p0, false, false);
        p1 = __builtin_amdgcn_wmma_f32_16x16x32_f16(false, a0, false, b10,
                                                    (short)0, p1, false, false);
        p0 = __builtin_amdgcn_wmma_f32_16x16x32_f16(false, a1, false, b01,
                                                    (short)0, p0, false, false);
        p1 = __builtin_amdgcn_wmma_f32_16x16x32_f16(false, a1, false, b11,
                                                    (short)0, p1, false, false);

        // reference clip, product accumulation (full-rate; packed muls)
#pragma unroll
        for (int r = 0; r < 8; ++r) {
            prod0[r] *= __builtin_amdgcn_fmed3f(p0[r], 1e-15f, 1e15f);
            prod1[r] *= __builtin_amdgcn_fmed3f(p1[r], 1e-15f, 1e15f);
        }
    }

    __syncthreads();   // table dead everywhere before aliased lsmem writes

    // ---- one raw v_log_f32 per element + bias (log2 domain) ---------------
    const float vba = ba[j], vbq = bq[j], vla = lama[j], vlq = lamq[j];
#pragma unroll
    for (int r = 0; r < 8; ++r) {
        int   l   = (lblk << 4) + (hi << 3) + r;    // C layout: M = r + hi*8
        float s   = (float)l * (1.0f / 64.0f);
        float dq  = s - vlq;
        float eB2 = (-vbq * dq * dq - vba * fabsf(s - vla)) * L2E;
        int col0  = (ib0 << 4) + lmod;              // C layout: N = lane%16
        lsmem[l * 64 + col0]      = __builtin_amdgcn_logf(prod0[r]) + eB2;
        lsmem[l * 64 + col0 + 16] = __builtin_amdgcn_logf(prod1[r]) + eB2;
    }
    __syncthreads();

    // ---- softmax over l (base-2). Thread c owns column c: address l*64+c
    // always hits bank c => conflict-free; 64 contiguous output floats.
    if (tid < 64) {
        float m = -3.0e38f;
#pragma unroll 8
        for (int l = 0; l < Q; ++l) m = fmaxf(m, lsmem[l * 64 + tid]);
        float ssum = 0.0f;
#pragma unroll 8
        for (int l = 0; l < Q; ++l) {
            float e = __builtin_amdgcn_exp2f(lsmem[l * 64 + tid] - m);
            ssum += e;
            lsmem[l * 64 + tid] = e;                // own column: no race
        }
        float inv = __builtin_amdgcn_rcpf(ssum);
        float* op = out + ((size_t)(i0 + tid) * F + j) * Q;
#pragma unroll
        for (int l = 0; l < Q; l += 4) {
            f32x4 v;
            v[0] = lsmem[(l + 0) * 64 + tid] * inv;
            v[1] = lsmem[(l + 1) * 64 + tid] * inv;
            v[2] = lsmem[(l + 2) * 64 + tid] * inv;
            v[3] = lsmem[(l + 3) * 64 + tid] * inv;
            *(f32x4*)(op + l) = v;
        }
    }
}

extern "C" void kernel_launch(void* const* d_in, const int* in_sizes, int n_in,
                              void* d_out, int out_size, void* d_ws, size_t ws_size,
                              hipStream_t stream) {
    (void)in_sizes; (void)n_in; (void)out_size;
    const float* x    = (const float*)d_in[0];
    const float* W    = (const float*)d_in[1];
    const float* ba   = (const float*)d_in[2];
    const float* bq   = (const float*)d_in[3];
    const float* lama = (const float*)d_in[4];
    const float* lamq = (const float*)d_in[5];
    float* out = (float*)d_out;

    dim3 grid(F, B_TOT / 64);                        // (j, batch tile)
    const size_t need = (size_t)B_TOT * F * Q * sizeof(_Float16);  // 2 MB
    if (d_ws != nullptr && ws_size >= need) {
        _Float16* xh = (_Float16*)d_ws;
        cvt_x_f16<<<(B_TOT * F * Q / 8 + 255) / 256, 256, 0, stream>>>(x, xh);
        drn_fused<true><<<grid, 256, 0, stream>>>(x, xh, W, ba, bq, lama, lamq, out);
    } else {
        drn_fused<false><<<grid, 256, 0, stream>>>(x, nullptr, W, ba, bq, lama, lamq, out);
    }
}
